// InfoPacker_63831803953660
// MI455X (gfx1250) — compile-verified, matrix operand order checked by reading, not run on hardware
//
#include <hip/hip_runtime.h>
#include <hip/hip_bf16.h>
#include <stdint.h>

#define B_ 2
#define F_ 256
#define D_ 768
#define N_ 12
#define H_ 64

typedef __attribute__((ext_vector_type(16))) _Float16 v16h;
typedef __attribute__((ext_vector_type(8)))  _Float16 v8h;
typedef __attribute__((ext_vector_type(8)))  float    v8f;
typedef __attribute__((ext_vector_type(4)))  unsigned int v4u;
typedef __attribute__((ext_vector_type(8)))  int      v8i;
typedef __attribute__((ext_vector_type(4)))  int      v4i;

#ifndef HAVE_TDM
#if defined(__has_builtin)
#if __has_builtin(__builtin_amdgcn_tensor_load_to_lds)
#define HAVE_TDM 1
#else
#define HAVE_TDM 0
#endif
#else
#define HAVE_TDM 0
#endif
#endif

// k offset of fragment element e (16-bit 16x32 A/B layout, ISA 7.12.2).
// e=0..7 -> k = 8*khalf + e ; e=8..15 -> k = 16 + 8*khalf + (e-8).
__device__ __forceinline__ int kmap(int e, int khalf) {
  int j = e >> 1;
  return ((j & 3) * 2 + (e & 1)) + ((j >> 2) * 16) + khalf * 8;
}

// A fragment from a row-major f16 row: two contiguous 16B chunks per lane.
__device__ __forceinline__ v16h load_a_f16(const _Float16* rowp, int khalf) {
  const _Float16* p = rowp + 8 * khalf;
  v8h lo = *(const v8h*)(p);
  v8h hi = *(const v8h*)(p + 16);
  return __builtin_shufflevector(lo, hi, 0, 1, 2, 3, 4, 5, 6, 7,
                                 8, 9, 10, 11, 12, 13, 14, 15);
}

// B fragment from a pre-packed fragment block: one 32B-aligned v16h per lane.
__device__ __forceinline__ v16h load_frag(const _Float16* fragblk, int lane) {
  return *(const v16h*)(fragblk + (lane << 4));
}

__device__ __forceinline__ v8f wmma16(v16h a, v16h b, v8f c) {
  return __builtin_amdgcn_wmma_f32_16x16x32_f16(false, a, false, b, (short)0, c,
                                                false, false);
}

__device__ __forceinline__ float sigm(float x) { return 1.0f / (1.0f + __expf(-x)); }

#if HAVE_TDM
// 2-D TDM load of f16 tiles: tile_d0 elems/row (contiguous), tile_d1 rows,
// row stride in elements, into LDS byte offset lds_off. data_size = 2B.
__device__ __forceinline__ void tdm_load_2d_f16(const _Float16* gptr, unsigned lds_off,
                                                unsigned tile_d0, unsigned tile_d1,
                                                unsigned row_stride) {
  unsigned long long ga = (unsigned long long)(uintptr_t)gptr;
  v4u g0 = { 1u, lds_off, (unsigned)(ga & 0xffffffffu),
             (unsigned)((ga >> 32) & 0x01ffffffu) | 0x80000000u };
  v8i g1 = { (int)(1u << 16),          // data_size = 2B
             (int)(tile_d0 << 16),     // tensor_dim0 lo16
             (int)(tile_d1 << 16),     // tensor_dim1 lo16
             (int)(tile_d0 << 16),     // tile_dim0
             (int)tile_d1,             // tile_dim1 (tile_dim2 = 0)
             (int)row_stride, 0, 0 };  // tensor_dim0_stride
  v4i gz = { 0, 0, 0, 0 };
#if __clang_major__ >= 23
  v8i gz8 = { 0, 0, 0, 0, 0, 0, 0, 0 };
  __builtin_amdgcn_tensor_load_to_lds(g0, g1, gz, gz, gz8, 0);
#else
  __builtin_amdgcn_tensor_load_to_lds(g0, g1, gz, gz, 0);
#endif
}
#endif

// ---------------------------------------------------------------------------
// Prep 0: fp32 -> f16 elementwise (for activations entering WMMA as A).
// ---------------------------------------------------------------------------
__global__ void cvt_f32_to_f16(const float* __restrict__ src,
                               _Float16* __restrict__ dst, int n) {
  int i = blockIdx.x * blockDim.x + threadIdx.x;
  if (i < n) dst[i] = (_Float16)src[i];
}

// ---------------------------------------------------------------------------
// Prep 1: repack a [K x N] fp32 matrix into per-lane B-fragment order:
// dst[((kb*(N/16)+ct)*32 + lane)*16 + e] = src[(kb*32 + kmap(e,lane>>4))*N + ct*16 + lane&15]
// ---------------------------------------------------------------------------
__global__ void repack_b(const float* __restrict__ src, _Float16* __restrict__ dst,
                         int K, int N) {
  int d = blockIdx.x * blockDim.x + threadIdx.x;
  if (d >= K * N) return;
  int e = d & 15;
  int lane = (d >> 4) & 31;
  int frag = d >> 9;
  int nt = N >> 4;
  int ct = frag % nt, kb = frag / nt;
  int k = kb * 32 + kmap(e, lane >> 4);
  int col = ct * 16 + (lane & 15);
  dst[d] = (_Float16)src[(size_t)k * N + col];
}

// ---------------------------------------------------------------------------
// 1) QKV projection: Xh[512x768] (f16) @ packed W -> q/k/v f16. 4 waves/block,
// wave tile 16x64, block tile 64x64.
// ---------------------------------------------------------------------------
__global__ void qkv_gemm(const _Float16* __restrict__ Xh,
                         const _Float16* __restrict__ Wvp, const float* __restrict__ bv,
                         const _Float16* __restrict__ Wqp, const float* __restrict__ bq,
                         const _Float16* __restrict__ Wkp, const float* __restrict__ bk,
                         _Float16* __restrict__ Vo, _Float16* __restrict__ Qo,
                         _Float16* __restrict__ Ko) {
  const int which = blockIdx.z;
  const _Float16* Wp = which == 0 ? Wvp : (which == 1 ? Wqp : Wkp);
  const float* bias = which == 0 ? bv : (which == 1 ? bq : bk);
  _Float16* O = which == 0 ? Vo : (which == 1 ? Qo : Ko);
  const int lane = threadIdx.x & 31, wave = threadIdx.x >> 5;
  const int khalf = lane >> 4, nc = lane & 15, m = lane & 15;
  const int row0 = blockIdx.x * 64 + wave * 16;
  const int col0 = blockIdx.y * 64;
  const int row = row0 + m;
  v8f acc[4] = {};
  for (int k0 = 0; k0 < D_; k0 += 32) {
    const int kb = k0 >> 5;
    v16h a = load_a_f16(Xh + (size_t)row * D_ + k0, khalf);
#pragma unroll
    for (int ct = 0; ct < 4; ++ct) {
      v16h b = load_frag(Wp + ((size_t)(kb * 48 + (col0 >> 4) + ct) << 9), lane);
      acc[ct] = wmma16(a, b, acc[ct]);
    }
  }
#pragma unroll
  for (int ct = 0; ct < 4; ++ct)
#pragma unroll
    for (int j = 0; j < 8; ++j) {
      int r = row0 + j + 8 * khalf;
      int c = col0 + ct * 16 + nc;
      O[(size_t)r * (N_ * H_) + c] = (_Float16)(acc[ct][j] + bias[c]);
    }
}

// ---------------------------------------------------------------------------
// 2) Attention per (b, head). Q,K f16 tiles (32KB each) TDM-loaded to LDS,
// V transpose-staged so P@V's B operand is a contiguous row read.
// Block 128 thr (4 waves); wave owns 4 f-tiles of 16 query rows.
// LDS: Qs 32KB | Ks 32KB | VsT 32KB | Sw fp32 64KB | Pw f16 32KB = 192KB.
// ---------------------------------------------------------------------------
__global__ void attn_kernel(const _Float16* __restrict__ qh,
                            const _Float16* __restrict__ kh,
                            const _Float16* __restrict__ vh,
                            const float* __restrict__ mask,
                            _Float16* __restrict__ attn_v) {
  extern __shared__ char smemraw[];
  _Float16* Qs  = (_Float16*)smemraw;                 // [256][64]
  _Float16* Ks  = (_Float16*)(smemraw + 32768);       // [256][64]
  _Float16* VsT = (_Float16*)(smemraw + 65536);       // [64][256]
  const int lane = threadIdx.x & 31, wave = threadIdx.x >> 5;
  float*    Sw  = (float*)(smemraw + 98304 + wave * 16384);            // [16][256]
  _Float16* Pw  = (_Float16*)(smemraw + 98304 + 65536 + wave * 8192);  // [16][256]
  const int bb = blockIdx.x / N_, head = blockIdx.x % N_;
  const int khalf = lane >> 4, nc = lane & 15, m = lane & 15;
  const size_t base = ((size_t)bb * F_ * N_ + head) * H_;  // [bb, f/t=0, head, 0]

#if HAVE_TDM
  if (wave == 0) {
    tdm_load_2d_f16(qh + base, 0u, 64u, 256u, (unsigned)(N_ * H_));
    tdm_load_2d_f16(kh + base, 32768u, 64u, 256u, (unsigned)(N_ * H_));
    __builtin_amdgcn_s_wait_tensorcnt((short)0);
  }
#else
  for (int idx = threadIdx.x; idx < 256 * 64; idx += blockDim.x) {
    int t = idx >> 6, h = idx & 63;
    size_t g = ((size_t)(bb * F_ + t) * N_ + head) * H_ + h;
    Qs[idx] = qh[g];
    Ks[idx] = kh[g];
  }
#endif
  // Transpose-stage V: VsT[h][t].
  for (int idx = threadIdx.x; idx < 256 * 64; idx += blockDim.x) {
    int h = idx & 63, t = idx >> 6;
    VsT[h * 256 + t] = vh[((size_t)(bb * F_ + t) * N_ + head) * H_ + h];
  }
  __syncthreads();

  for (int ft = wave; ft < 16; ft += 4) {
    const int f0 = ft * 16;
    // S = Q K^T : 16 t-tiles, K-dim = 64 over h.
    for (int tt = 0; tt < 16; ++tt) {
      const int t0 = tt * 16;
      v8f acc = {};
#pragma unroll
      for (int k0 = 0; k0 < 64; k0 += 32) {
        v16h a = load_a_f16(Qs + (f0 + m) * 64 + k0, khalf);
        v16h b = load_a_f16(Ks + (t0 + nc) * 64 + k0, khalf);  // B[k][t]=K[t][k]
        acc = wmma16(a, b, acc);
      }
#pragma unroll
      for (int j = 0; j < 8; ++j) {
        int r = j + 8 * khalf;
        float mk = mask[((size_t)bb * F_ + f0 + r) * F_ + t0 + nc];
        Sw[r * 256 + t0 + nc] = acc[j] * 0.125f + (1.0f - mk) * -10000.0f;
      }
    }
    // fp32 row softmax (8 vals/lane, wave32 shfl reductions) -> f16 probs.
    for (int r = 0; r < 16; ++r) {
      float mx = -3.4e38f;
#pragma unroll
      for (int i = 0; i < 8; ++i) mx = fmaxf(mx, Sw[r * 256 + lane + 32 * i]);
      for (int off = 16; off > 0; off >>= 1) mx = fmaxf(mx, __shfl_xor(mx, off, 32));
      float p[8], sum = 0.0f;
#pragma unroll
      for (int i = 0; i < 8; ++i) {
        p[i] = __expf(Sw[r * 256 + lane + 32 * i] - mx);
        sum += p[i];
      }
      for (int off = 16; off > 0; off >>= 1) sum += __shfl_xor(sum, off, 32);
      float inv = 1.0f / sum;
#pragma unroll
      for (int i = 0; i < 8; ++i)
        Pw[r * 256 + lane + 32 * i] = (_Float16)(p[i] * inv);
    }
    // O = P[16x256] @ V[256x64]; B from VsT rows (contiguous).
#pragma unroll
    for (int ct = 0; ct < 4; ++ct) {
      const int i0 = ct * 16;
      v8f acc = {};
      for (int k0 = 0; k0 < 256; k0 += 32) {
        v16h a = load_a_f16(Pw + m * 256 + k0, khalf);
        v16h b = load_a_f16(VsT + (i0 + nc) * 256 + k0, khalf);
        acc = wmma16(a, b, acc);
      }
#pragma unroll
      for (int j = 0; j < 8; ++j) {
        int r = j + 8 * khalf;
        attn_v[((size_t)(bb * F_ + f0 + r) * N_ + head) * H_ + i0 + nc] =
            (_Float16)acc[j];
      }
    }
  }
}

// ---------------------------------------------------------------------------
// 3) BiLSTM: block per direction, 256 thr (8 waves). Pre-packed [Wx;Wh]
// fragments (64KB) staged to LDS; per step z = [x h] @ Wcat via WMMA
// (ds_load_b128 fragments), fp32 gates, h stored f16 at true time index.
// LDS: Wls 64KB | Als f16 8KB | Zls fp32 32KB | Cls fp32 6KB = 110KB.
// ---------------------------------------------------------------------------
__global__ void lstm_kernel(const _Float16* __restrict__ vh,
                            const _Float16* __restrict__ Wf_p,
                            const float* __restrict__ b_f,
                            const _Float16* __restrict__ Wb_p,
                            const float* __restrict__ b_b,
                            _Float16* __restrict__ hs_f, _Float16* __restrict__ hs_b) {
  extern __shared__ char smemraw[];
  _Float16* Wls = (_Float16*)smemraw;                  // packed frags [128x256]
  _Float16* Als = (_Float16*)(smemraw + 65536);        // [32][128]
  float*    Zls = (float*)(smemraw + 65536 + 8192);    // [32][256]
  float*    Cls = (float*)(smemraw + 65536 + 8192 + 32768);  // [24][64]
  const int dir = blockIdx.x;
  const _Float16* Wp = dir ? Wb_p : Wf_p;
  const float* bias = dir ? b_b : b_f;
  _Float16* hs = dir ? hs_b : hs_f;
  const int tid = threadIdx.x, lane = tid & 31, wave = tid >> 5;
  const int khalf = lane >> 4, nc = lane & 15;

  for (int i = tid; i < 128 * 256; i += blockDim.x) Wls[i] = Wp[i];
  for (int i = tid; i < 32 * 128; i += blockDim.x) Als[i] = (_Float16)0.0f;
  for (int i = tid; i < 24 * 64; i += blockDim.x) Cls[i] = 0.0f;
  __syncthreads();

  for (int s = 0; s < F_; ++s) {
    const int t = dir ? (F_ - 1 - s) : s;
    for (int i = tid; i < 24 * 64; i += blockDim.x) {   // x_t rows = (b, head)
      int r = i >> 6, h = i & 63;
      int b2 = r / N_, nh = r % N_;
      Als[r * 128 + h] = vh[((size_t)(b2 * F_ + t) * N_ + nh) * H_ + h];
    }
    __syncthreads();
#pragma unroll
    for (int tile = 0; tile < 4; ++tile) {              // 32 tiles over 8 waves
      int tid2 = wave + tile * 8;
      int r0 = (tid2 >> 4) * 16, c0t = tid2 & 15;
      v8f acc = {};
#pragma unroll
      for (int kb = 0; kb < 4; ++kb) {
        v16h a = load_a_f16(Als + (r0 + (lane & 15)) * 128 + kb * 32, khalf);
        v16h b = load_frag(Wls + ((size_t)(kb * 16 + c0t) << 9), lane);
        acc = wmma16(a, b, acc);
      }
#pragma unroll
      for (int j = 0; j < 8; ++j)
        Zls[(r0 + j + 8 * khalf) * 256 + c0t * 16 + nc] = acc[j] + bias[c0t * 16 + nc];
    }
    __syncthreads();
    for (int i = tid; i < 24 * 64; i += blockDim.x) {   // gates i,f,g,o (fp32)
      int r = i >> 6, h = i & 63;
      float zi = Zls[r * 256 + h],       zf = Zls[r * 256 + 64 + h];
      float zg = Zls[r * 256 + 128 + h], zo = Zls[r * 256 + 192 + h];
      float c = sigm(zf) * Cls[i] + sigm(zi) * tanhf(zg);
      float hh = sigm(zo) * tanhf(c);
      Cls[i] = c;
      Als[r * 128 + 64 + h] = (_Float16)hh;
      hs[((size_t)t * 24 + r) * 64 + h] = (_Float16)hh;
    }
    __syncthreads();
  }
}

// ---------------------------------------------------------------------------
// 4) pure_ctx: [6144 x 128] @ packed [128 x 64] + bias.
// ---------------------------------------------------------------------------
__global__ void pure_ctx_kernel(const _Float16* __restrict__ hs_f,
                                const _Float16* __restrict__ hs_b,
                                const _Float16* __restrict__ Pp,
                                const float* __restrict__ pb,
                                _Float16* __restrict__ pure) {
  const int lane = threadIdx.x & 31, wave = threadIdx.x >> 5;
  const int khalf = lane >> 4, nc = lane & 15, m = lane & 15;
  const int row0 = blockIdx.x * 64 + wave * 16;
  const int row = row0 + m;
  const int b2 = row / (F_ * N_), f = (row / N_) % F_, nh = row % N_;
  const int seqr = b2 * N_ + nh;
  v8f acc[4] = {};
#pragma unroll
  for (int kb = 0; kb < 4; ++kb) {
    const int k0 = kb * 32;
    const _Float16* src = (k0 < 64)
        ? (hs_f + ((size_t)f * 24 + seqr) * 64 + k0)
        : (hs_b + ((size_t)f * 24 + seqr) * 64 + (k0 - 64));
    v16h a = load_a_f16(src, khalf);
#pragma unroll
    for (int ct = 0; ct < 4; ++ct) {
      v16h b = load_frag(Pp + ((size_t)(kb * 4 + ct) << 9), lane);
      acc[ct] = wmma16(a, b, acc[ct]);
    }
  }
#pragma unroll
  for (int ct = 0; ct < 4; ++ct)
#pragma unroll
    for (int j = 0; j < 8; ++j)
      pure[(size_t)(row0 + j + 8 * khalf) * 64 + ct * 16 + nc] =
          (_Float16)(acc[ct][j] + pb[ct * 16 + nc]);
}

// ---------------------------------------------------------------------------
// 5) directed_ctx: [6144 x 4096] @ packed builder [4096 x 64]; A-row =
// kron(q_row, attn_v_row) built on the fly in f16 (no 100MB 5-D tensor).
// Block 128 thr; 16 rows/block; wave w -> output cols w*16..w*16+15.
// ---------------------------------------------------------------------------
__global__ void directed_kernel(const _Float16* __restrict__ qh,
                                const _Float16* __restrict__ avh,
                                const _Float16* __restrict__ builderp,
                                const float* __restrict__ dbias,
                                _Float16* __restrict__ directed) {
  __shared__ _Float16 qs[16 * 64];
  __shared__ _Float16 as[16 * 64];
  const int lane = threadIdx.x & 31, wave = threadIdx.x >> 5;
  const int khalf = lane >> 4, nc = lane & 15, m = lane & 15;
  const int r0 = blockIdx.x * 16;
  for (int i = threadIdx.x; i < 16 * 64; i += blockDim.x) {
    qs[i] = qh[(size_t)r0 * 64 + i];
    as[i] = avh[(size_t)r0 * 64 + i];
  }
  __syncthreads();
  v8f acc = {};
  for (int kb = 0; kb < 128; ++kb) {
    const int k0 = kb * 32;
    v16h a;
#pragma unroll
    for (int e = 0; e < 16; ++e) {
      int k = k0 + kmap(e, khalf);                      // k = kq*64 + kh
      a[e] = qs[m * 64 + (k >> 6)] * as[m * 64 + (k & 63)];
    }
    v16h b = load_frag(builderp + ((size_t)(kb * 4 + wave) << 9), lane);
    acc = wmma16(a, b, acc);
  }
#pragma unroll
  for (int j = 0; j < 8; ++j)
    directed[(size_t)(r0 + j + 8 * khalf) * 64 + wave * 16 + nc] =
        (_Float16)(acc[j] + dbias[wave * 16 + nc]);
}

// ---------------------------------------------------------------------------
// 6) Output: [512 x 2304] @ packed [2304 x 768] + bias -> fp32 out.
// A gathered from v/pure/directed f16 segments (all row-contiguous).
// ---------------------------------------------------------------------------
__global__ void output_gemm(const _Float16* __restrict__ vh,
                            const _Float16* __restrict__ pure,
                            const _Float16* __restrict__ directed,
                            const _Float16* __restrict__ Woutp,
                            const float* __restrict__ ob,
                            float* __restrict__ out) {
  const int lane = threadIdx.x & 31, wave = threadIdx.x >> 5;
  const int khalf = lane >> 4, nc = lane & 15, m = lane & 15;
  const int row0 = blockIdx.x * 64 + wave * 16;
  const int col0 = blockIdx.y * 64;
  const int row = row0 + m;
  v8f acc[4] = {};
  for (int kb = 0; kb < 72; ++kb) {
    const int k0 = kb * 32;
    const int n = k0 / 192, x0 = k0 % 192;              // x = [v | pure | directed]
    const _Float16* seg = (x0 < 64) ? vh : (x0 < 128 ? pure : directed);
    v16h a = load_a_f16(seg + ((size_t)row * N_ + n) * 64 + (x0 & 63), khalf);
#pragma unroll
    for (int ct = 0; ct < 4; ++ct) {
      v16h b = load_frag(Woutp + ((size_t)(kb * 48 + (col0 >> 4) + ct) << 9), lane);
      acc[ct] = wmma16(a, b, acc[ct]);
    }
  }
#pragma unroll
  for (int ct = 0; ct < 4; ++ct)
#pragma unroll
    for (int j = 0; j < 8; ++j) {
      int r = row0 + j + 8 * khalf;
      int c = col0 + ct * 16 + nc;
      out[(size_t)r * D_ + c] = acc[ct][j] + ob[c];
    }
}

// ---------------------------------------------------------------------------
extern "C" void kernel_launch(void* const* d_in, const int* in_sizes, int n_in,
                              void* d_out, int out_size, void* d_ws, size_t ws_size,
                              hipStream_t stream) {
  (void)in_sizes; (void)n_in; (void)out_size; (void)ws_size;
  const float* X       = (const float*)d_in[0];
  const float* builder = (const float*)d_in[1];
  const float* mask    = (const float*)d_in[2];
  const float* Wv = (const float*)d_in[3];   const float* bv = (const float*)d_in[4];
  const float* Wq = (const float*)d_in[5];   const float* bq = (const float*)d_in[6];
  const float* Wk = (const float*)d_in[7];   const float* bk = (const float*)d_in[8];
  const float* Pk = (const float*)d_in[9];   const float* Pb = (const float*)d_in[10];
  const float* Wo = (const float*)d_in[11];  const float* ob = (const float*)d_in[12];
  const float* db = (const float*)d_in[13];
  const float* Wxf = (const float*)d_in[14]; const float* Whf = (const float*)d_in[15];
  const float* bf  = (const float*)d_in[16];
  const float* Wxb = (const float*)d_in[17]; const float* Whb = (const float*)d_in[18];
  const float* bbw = (const float*)d_in[19];

  // f16 workspace carve (all counts multiples of 16 -> 32B aligned).
  _Float16* hb = (_Float16*)d_ws;
  size_t o = 0;
  _Float16* Xh   = hb + o; o += 393216;       // [512][768]
  _Float16* Wvp  = hb + o; o += 589824;       // packed 768x768
  _Float16* Wqp  = hb + o; o += 589824;
  _Float16* Wkp  = hb + o; o += 589824;
  _Float16* Bp   = hb + o; o += 262144;       // packed builder 4096x64
  _Float16* Pp   = hb + o; o += 8192;         // packed 128x64
  _Float16* Wop  = hb + o; o += 1769472;      // packed 2304x768
  _Float16* Wfp  = hb + o; o += 32768;        // packed [Wx;Wh] fwd 128x256
  _Float16* Wbp  = hb + o; o += 32768;        // packed [Wx;Wh] bwd
  _Float16* q    = hb + o; o += 393216;       // activations (f16)
  _Float16* k    = hb + o; o += 393216;
  _Float16* v    = hb + o; o += 393216;
  _Float16* av   = hb + o; o += 393216;
  _Float16* hsf  = hb + o; o += 393216;       // [F][24][64]
  _Float16* hsb  = hb + o; o += 393216;
  _Float16* pure = hb + o; o += 393216;
  _Float16* dctx = hb + o; o += 393216;
  float* out = (float*)d_out;

  // Prep: convert X, pack all weights into WMMA B-fragment order.
  cvt_f32_to_f16<<<dim3(1536), 256, 0, stream>>>(X, Xh, 393216);
  repack_b<<<dim3(2304), 256, 0, stream>>>(Wv, Wvp, 768, 768);
  repack_b<<<dim3(2304), 256, 0, stream>>>(Wq, Wqp, 768, 768);
  repack_b<<<dim3(2304), 256, 0, stream>>>(Wk, Wkp, 768, 768);
  repack_b<<<dim3(1024), 256, 0, stream>>>(builder, Bp, 4096, 64);
  repack_b<<<dim3(32),   256, 0, stream>>>(Pk, Pp, 128, 64);
  repack_b<<<dim3(6912), 256, 0, stream>>>(Wo, Wop, 2304, 768);
  repack_b<<<dim3(64),   256, 0, stream>>>(Wxf, Wfp, 64, 256);
  repack_b<<<dim3(64),   256, 0, stream>>>(Whf, Wfp + 16384, 64, 256);
  repack_b<<<dim3(64),   256, 0, stream>>>(Wxb, Wbp, 64, 256);
  repack_b<<<dim3(64),   256, 0, stream>>>(Whb, Wbp + 16384, 64, 256);

  qkv_gemm<<<dim3(8, 12, 3), 128, 0, stream>>>(Xh, Wvp, bv, Wqp, bq, Wkp, bk, v, q, k);
  attn_kernel<<<dim3(B_ * N_), 128, 196608, stream>>>(q, k, v, mask, av);
  lstm_kernel<<<dim3(2), 256, 112640, stream>>>(v, Wfp, bf, Wbp, bbw, hsf, hsb);
  pure_ctx_kernel<<<dim3(96), 128, 0, stream>>>(hsf, hsb, Pp, Pb, pure);
  directed_kernel<<<dim3(384), 128, 0, stream>>>(q, av, Bp, db, dctx);
  output_gemm<<<dim3(8, 12), 128, 0, stream>>>(v, pure, dctx, Wop, ob, out);
}
